// DeepSeekV3_8589934592344
// MI455X (gfx1250) — matrix operand order repaired and checked by reading, not compiled, not run on hardware
//
#include <hip/hip_runtime.h>

// ---------------- problem constants (from reference) ----------------
#define S_    512
#define H_    1024
#define NH_   16
#define HD_   64
#define HD2_  32
#define CKV_  256
#define CQ_   384
#define E_    8
#define L_    2
#define F_    4096
#define V_    32000
#define NHHD_ 1024

// ---------------- WMMA types ----------------
typedef float  v8f   __attribute__((ext_vector_type(8)));
typedef __bf16 v16bf __attribute__((ext_vector_type(16)));
typedef __bf16 v8bf  __attribute__((ext_vector_type(8)));
typedef int    v4i   __attribute__((ext_vector_type(4)));

#define GF_GELU 1
#define GF_ACC  2

// Opportunistic CDNA5 async global->LDS copy path
#if defined(__has_builtin)
#if __has_builtin(__builtin_amdgcn_global_load_async_to_lds_b128) && \
    __has_builtin(__builtin_amdgcn_s_wait_asynccnt)
#define USE_ASYNC_COPY 1
#endif
#endif

__device__ __forceinline__ float gelu_exact(float x) {
  return 0.5f * x * (1.0f + erff(x * 0.70710678118654752f));
}

__device__ __forceinline__ v16bf cat8(v8bf lo, v8bf hi) {
  return __builtin_shufflevector(lo, hi, 0, 1, 2, 3, 4, 5, 6, 7,
                                 8, 9, 10, 11, 12, 13, 14, 15);
}

// ---------------- generic batched WMMA GEMM ----------------
// C[b] = epilogue( alpha * A[b] @ B[b] )
// Tiles: BM=256, BN=64, BK=32. 256 threads = 8 waves (4 in M x 2 in N),
// each wave computes a 64x32 output via 4x2 v_wmma_f32_16x16x32_bf16.
// Pipeline per K step: async f32 tile -> LDS stage; one convert pass
// (f32 LDS -> bf16 LDS, B transposed to [N][K]); fragments are then pure
// contiguous b128 DS loads. Next tile's async copy overlaps WMMA.
#define BM 256
#define BN 64
#define BK 32
#define LDAF 36   // f32 A stage row stride (dwords)
#define LDBF 68   // f32 B stage row stride (dwords)
#define LDAB 40   // bf16 A tile row stride (elements), [M][K]
#define LDBB 40   // bf16 B tile row stride (elements), [N][K]

__global__ __launch_bounds__(256) void k_gemm(
    const float* __restrict__ A, int lda, long long sA,
    const float* __restrict__ B, int ldb, long long sB,
    const float* __restrict__ bias,
    const float* __restrict__ residual,
    const float* __restrict__ rowscale, int rsStride,
    float* __restrict__ C, int ldc, long long sC,
    int M, int N, int K, float alpha, int flags)
{
  __shared__ float  lAf[BM * LDAF];   // 36 KB f32 stage
  __shared__ float  lBf[BK * LDBF];   // 8.5 KB f32 stage
  __shared__ __bf16 lAb[BM * LDAB];   // 20 KB bf16 [M][K]
  __shared__ __bf16 lBb[BN * LDBB];   // 5 KB bf16 [N][K] (transposed)

  const int tid  = threadIdx.x;
  const int lane = tid & 31;
  const int wave = tid >> 5;
  const int wm   = wave & 3;   // wave row block (0..3) -> 64 rows each
  const int wn   = wave >> 2;  // wave col block (0..1) -> 32 cols each
  const int hs   = lane >> 4;  // lane half select
  const int l15  = lane & 15;

  const int rowBase = blockIdx.y * BM;
  const int colBase = blockIdx.x * BN;
  const int bz = blockIdx.z;

  const float* Ab = A + (long long)bz * sA;
  const float* Bb = B + (long long)bz * sB;
  float*       Cb = C + (long long)bz * sC;

  v8f acc[4][2] = {};

#ifdef USE_ASYNC_COPY
  auto issue_async = [&](int k0) {
    // A tile: 256x32 f32 = 2048 16B chunks, 8 per thread
#pragma unroll
    for (int i = 0; i < 8; ++i) {
      int idx = tid + i * 256;
      int r = idx >> 3;
      int c4 = (idx & 7) << 2;
      __builtin_amdgcn_global_load_async_to_lds_b128(
          (v4i*)(Ab + (long long)(rowBase + r) * lda + k0 + c4),
          (v4i*)(&lAf[r * LDAF + c4]), 0, 0);
    }
    // B tile: 32x64 f32 = 512 16B chunks, 2 per thread
#pragma unroll
    for (int i = 0; i < 2; ++i) {
      int idx = tid + i * 256;
      int r = idx >> 4;
      int c4 = (idx & 15) << 2;
      __builtin_amdgcn_global_load_async_to_lds_b128(
          (v4i*)(Bb + (long long)(k0 + r) * ldb + colBase + c4),
          (v4i*)(&lBf[r * LDBF + c4]), 0, 0);
    }
    // prefetch one tile further into L2 (global_prefetch_b8)
    if (k0 + BK < K) {
      __builtin_prefetch(Ab + (long long)(rowBase + tid) * lda + (k0 + BK), 0, 1);
      if (tid < BK)
        __builtin_prefetch(Bb + (long long)(k0 + BK + tid) * ldb + colBase, 0, 1);
    }
  };
#endif

  auto convert_pass = [&]() {
    // A: one 32-wide row per thread, f32 -> bf16, contiguous b128 stores
    const float* srcA = &lAf[tid * LDAF];
#pragma unroll
    for (int g = 0; g < 4; ++g) {
      v8bf t;
#pragma unroll
      for (int j = 0; j < 8; ++j) t[j] = (__bf16)srcA[g * 8 + j];
      *(v8bf*)&lAb[tid * LDAB + g * 8] = t;
    }
    // B: transpose [K][N] f32 -> [N][K] bf16; each thread one 8-K sliver
    const int c  = tid >> 2;
    const int kq = (tid & 3) << 3;
    v8bf tb;
#pragma unroll
    for (int j = 0; j < 8; ++j) tb[j] = (__bf16)lBf[(kq + j) * LDBF + c];
    *(v8bf*)&lBb[c * LDBB + kq] = tb;
  };

  auto do_wmma = [&]() {
    // A frag (16x32): lanes 0-15: K{0..7,16..23}; lanes 16-31: K{8..15,24..31}
    v16bf af[4];
#pragma unroll
    for (int mi = 0; mi < 4; ++mi) {
      const int r = wm * 64 + mi * 16 + l15;
      v8bf lo = *(const v8bf*)&lAb[r * LDAB + hs * 8];
      v8bf hi = *(const v8bf*)&lAb[r * LDAB + 16 + hs * 8];
      af[mi] = cat8(lo, hi);
    }
    // B frag (32x16): lanes 0-15: col N, K{0..7,16..23}; lanes 16-31: K{8..15,24..31}
    v16bf bfr[2];
#pragma unroll
    for (int ni = 0; ni < 2; ++ni) {
      const int c = wn * 32 + ni * 16 + l15;
      v8bf lo = *(const v8bf*)&lBb[c * LDBB + hs * 8];
      v8bf hi = *(const v8bf*)&lBb[c * LDBB + 16 + hs * 8];
      bfr[ni] = cat8(lo, hi);
    }
#pragma unroll
    for (int mi = 0; mi < 4; ++mi)
#pragma unroll
      for (int ni = 0; ni < 2; ++ni)
        acc[mi][ni] = __builtin_amdgcn_wmma_f32_16x16x32_bf16(
            false, af[mi], false, bfr[ni], (short)0, acc[mi][ni], false, false);
  };

#ifdef USE_ASYNC_COPY
  issue_async(0);
  __builtin_amdgcn_s_wait_asynccnt(0);
  __syncthreads();
  for (int k0 = 0; k0 < K; k0 += BK) {
    convert_pass();
    __syncthreads();                       // bf16 tiles ready; f32 stage consumed
    if (k0 + BK < K) issue_async(k0 + BK); // async DMA overlaps WMMA
    do_wmma();
    if (k0 + BK < K) {
      __builtin_amdgcn_s_wait_asynccnt(0);
      __syncthreads();                     // WMMA reads done + new f32 stage landed
    }
  }
#else
  for (int k0 = 0; k0 < K; k0 += BK) {
#pragma unroll
    for (int i = 0; i < 8; ++i) {
      int idx = tid + i * 256;
      int r = idx >> 3;
      int c4 = (idx & 7) << 2;
      const float4 va = *(const float4*)(Ab + (long long)(rowBase + r) * lda + k0 + c4);
      *(float4*)&lAf[r * LDAF + c4] = va;
    }
#pragma unroll
    for (int i = 0; i < 2; ++i) {
      int idx = tid + i * 256;
      int r = idx >> 4;
      int c4 = (idx & 15) << 2;
      const float4 vbv = *(const float4*)(Bb + (long long)(k0 + r) * ldb + colBase + c4);
      *(float4*)&lBf[r * LDBF + c4] = vbv;
    }
    if (k0 + BK < K) {
      __builtin_prefetch(Ab + (long long)(rowBase + tid) * lda + (k0 + BK), 0, 1);
      if (tid < BK)
        __builtin_prefetch(Bb + (long long)(k0 + BK + tid) * ldb + colBase, 0, 1);
    }
    __syncthreads();
    convert_pass();
    __syncthreads();
    do_wmma();
    __syncthreads();
  }
#endif

  // ---- epilogue ----
  // C/D layout: lanes 0-15: N=lane, M=j; lanes 16-31: N=lane-16, M=j+8
#pragma unroll
  for (int mi = 0; mi < 4; ++mi) {
#pragma unroll
    for (int ni = 0; ni < 2; ++ni) {
      const int gn = colBase + wn * 32 + ni * 16 + l15;
#pragma unroll
      for (int j = 0; j < 8; ++j) {
        const int gm = rowBase + wm * 64 + mi * 16 + hs * 8 + j;
        float v = acc[mi][ni][j] * alpha;
        if (bias) v += bias[gn];
        if (flags & GF_GELU) v = gelu_exact(v);
        if (rowscale) v *= rowscale[(long long)gm * rsStride];
        const long long ci = (long long)gm * ldc + gn;
        if (residual) v += residual[ci];
        if (flags & GF_ACC) v += Cb[ci];
        Cb[ci] = v;
      }
    }
  }
}

// ---------------- embedding gather ----------------
__global__ __launch_bounds__(256) void k_embed(const int* __restrict__ ids,
                                               const float* __restrict__ emb,
                                               float* __restrict__ x)
{
  const int t = blockIdx.x;
  const int row = ids[t];
  const float4* src = (const float4*)(emb + (long long)row * H_);
  float4* dst = (float4*)(x + (long long)t * H_);
  for (int i = threadIdx.x; i < H_ / 4; i += 256) dst[i] = src[i];
}

// ---------------- LayerNorm (one block per row) ----------------
__global__ __launch_bounds__(256) void k_ln(const float* __restrict__ x,
                                            const float* __restrict__ g,
                                            const float* __restrict__ b,
                                            float* __restrict__ y, int Hd)
{
  __shared__ float s1[256], s2[256];
  const int r = blockIdx.x, tid = threadIdx.x;
  const float* xr = x + (long long)r * Hd;
  float a = 0.f, q = 0.f;
  for (int i = tid; i < Hd; i += 256) { float v = xr[i]; a += v; q += v * v; }
  s1[tid] = a; s2[tid] = q; __syncthreads();
  for (int s = 128; s > 0; s >>= 1) {
    if (tid < s) { s1[tid] += s1[tid + s]; s2[tid] += s2[tid + s]; }
    __syncthreads();
  }
  const float mean = s1[0] / Hd;
  const float var  = s2[0] / Hd - mean * mean;
  const float inv  = rsqrtf(var + 1e-5f);
  for (int i = tid; i < Hd; i += 256)
    y[(long long)r * Hd + i] = (xr[i] - mean) * inv * g[i] + b[i];
}

// ---------------- RoPE for q (in place) and k -> kT [NH][HD][S] ----------------
__global__ __launch_bounds__(256) void k_rope(float* __restrict__ q,
                                              const float* __restrict__ k,
                                              float* __restrict__ kT)
{
  const int idx = blockIdx.x * 256 + threadIdx.x;       // S*NH*HD2 threads
  const int i = idx % HD2_;
  const int h = (idx / HD2_) % NH_;
  const int s = idx / (HD2_ * NH_);
  const float inv = __powf(10000.0f, -(float)(2 * i) / (float)HD_);
  const float ang = (float)s * inv;
  const float sn = __sinf(ang), cs = __cosf(ang);
  const long long base = ((long long)s * NH_ + h) * HD_ + 2 * i;
  const float q1 = q[base], q2 = q[base + 1];
  q[base]     = q1 * cs - q2 * sn;
  q[base + 1] = q1 * sn + q2 * cs;
  const float k1 = k[base], k2 = k[base + 1];
  kT[((long long)h * HD_ + 2 * i)     * S_ + s] = k1 * cs - k2 * sn;
  kT[((long long)h * HD_ + 2 * i + 1) * S_ + s] = k1 * sn + k2 * cs;
}

// ---------------- causal masked softmax over scores [NH][S][S] ----------------
__global__ __launch_bounds__(256) void k_softmax(float* __restrict__ sc)
{
  const int qr = blockIdx.x, h = blockIdx.y, tid = threadIdx.x;
  float* row = sc + ((long long)h * S_ + qr) * S_;
  const int n = qr + 1;
  __shared__ float red[256];
  float mx = -3.0e38f;
  for (int j = tid; j < n; j += 256) mx = fmaxf(mx, row[j]);
  red[tid] = mx; __syncthreads();
  for (int s = 128; s > 0; s >>= 1) {
    if (tid < s) red[tid] = fmaxf(red[tid], red[tid + s]);
    __syncthreads();
  }
  mx = red[0]; __syncthreads();
  float sum = 0.f;
  for (int j = tid; j < n; j += 256) { float e = __expf(row[j] - mx); row[j] = e; sum += e; }
  red[tid] = sum; __syncthreads();
  for (int s = 128; s > 0; s >>= 1) {
    if (tid < s) red[tid] += red[tid + s];
    __syncthreads();
  }
  const float inv = 1.0f / red[0];
  for (int j = tid; j < n; j += 256) row[j] *= inv;
  for (int j = n + tid; j < S_; j += 256) row[j] = 0.f;
}

// ---------------- gate: logits + softmax + top-2 -> dense wtok [S][E] ----------------
__global__ __launch_bounds__(256) void k_gate(const float* __restrict__ xf,
                                              const float* __restrict__ gW,
                                              const float* __restrict__ gb,
                                              const float* __restrict__ gbias,
                                              float* __restrict__ wtok)
{
  __shared__ float red[256 * E_];
  const int t = blockIdx.x, tid = threadIdx.x;
  float acc[E_];
#pragma unroll
  for (int e = 0; e < E_; ++e) acc[e] = 0.f;
  for (int i = tid; i < H_; i += 256) {
    const float xv = xf[(long long)t * H_ + i];
#pragma unroll
    for (int e = 0; e < E_; ++e) acc[e] += xv * gW[i * E_ + e];
  }
#pragma unroll
  for (int e = 0; e < E_; ++e) red[tid * E_ + e] = acc[e];
  __syncthreads();
  for (int s = 128; s > 0; s >>= 1) {
    if (tid < s) {
#pragma unroll
      for (int e = 0; e < E_; ++e) red[tid * E_ + e] += red[(tid + s) * E_ + e];
    }
    __syncthreads();
  }
  if (tid == 0) {
    float p[E_]; float mx = -3.0e38f;
#pragma unroll
    for (int e = 0; e < E_; ++e) { p[e] = red[e] + gb[e] + gbias[e]; mx = fmaxf(mx, p[e]); }
    float sum = 0.f;
#pragma unroll
    for (int e = 0; e < E_; ++e) { p[e] = __expf(p[e] - mx); sum += p[e]; }
#pragma unroll
    for (int e = 0; e < E_; ++e) p[e] /= sum;
    int i0 = 0;
#pragma unroll
    for (int e = 1; e < E_; ++e) if (p[e] > p[i0]) i0 = e;
    int i1 = (i0 == 0) ? 1 : 0;
#pragma unroll
    for (int e = 0; e < E_; ++e) if (e != i0 && p[e] > p[i1]) i1 = e;
    const float s2 = p[i0] + p[i1];
#pragma unroll
    for (int e = 0; e < E_; ++e)
      wtok[t * E_ + e] = (e == i0) ? p[i0] / s2 : ((e == i1) ? p[i1] / s2 : 0.f);
  }
}

// ---------------- host side ----------------
static inline void GEMM(const float* A, int lda, long long sA,
                        const float* B, int ldb, long long sB,
                        const float* bias, const float* residual,
                        const float* rowscale, int rsStride,
                        float* C, int ldc, long long sC,
                        int M, int N, int K, int batch, float alpha, int flags,
                        hipStream_t stream)
{
  dim3 grid(N / BN, M / BM, batch);
  k_gemm<<<grid, 256, 0, stream>>>(A, lda, sA, B, ldb, sB, bias, residual,
                                   rowscale, rsStride, C, ldc, sC, M, N, K, alpha, flags);
}

extern "C" void kernel_launch(void* const* d_in, const int* in_sizes, int n_in,
                              void* d_out, int out_size, void* d_ws, size_t ws_size,
                              hipStream_t stream)
{
  (void)in_sizes; (void)n_in; (void)out_size; (void)ws_size;
  const int*   ids   = (const int*)d_in[0];
  /* d_in[1] = attn_mask (causal tril) handled analytically */
  const float* emb   = (const float*)d_in[2];
  const float* ln1g  = (const float*)d_in[3];
  const float* ln1b  = (const float*)d_in[4];
  const float* Wdkv  = (const float*)d_in[5];
  const float* bdkv  = (const float*)d_in[6];
  const float* Wuk   = (const float*)d_in[7];
  const float* buk   = (const float*)d_in[8];
  const float* Wuv   = (const float*)d_in[9];
  const float* buv   = (const float*)d_in[10];
  const float* Wdq   = (const float*)d_in[11];
  const float* bdq   = (const float*)d_in[12];
  const float* Wuq   = (const float*)d_in[13];
  const float* buq   = (const float*)d_in[14];
  const float* Wo    = (const float*)d_in[15];
  const float* bo    = (const float*)d_in[16];
  const float* ln2g  = (const float*)d_in[17];
  const float* ln2b  = (const float*)d_in[18];
  const float* gateW = (const float*)d_in[19];
  const float* gateb = (const float*)d_in[20];
  const float* gatebias = (const float*)d_in[21];
  const float* sW1   = (const float*)d_in[22];
  const float* sb1   = (const float*)d_in[23];
  const float* sW2   = (const float*)d_in[24];
  const float* sb2   = (const float*)d_in[25];
  const float* eW1   = (const float*)d_in[26];
  const float* eb1   = (const float*)d_in[27];
  const float* eW2   = (const float*)d_in[28];
  const float* eb2   = (const float*)d_in[29];
  const float* flng  = (const float*)d_in[30];
  const float* flnb  = (const float*)d_in[31];
  const float* outW  = (const float*)d_in[32];
  const float* outb  = (const float*)d_in[33];
  float* out = (float*)d_out;

  // workspace carve (f32, all 16B aligned)
  float* p = (float*)d_ws;
  float* x0 = p;  p += S_ * H_;
  float* x1 = p;  p += S_ * H_;
  float* x2 = p;  p += S_ * H_;
  float* hb = p;  p += S_ * H_;
  float* kvb = p; p += S_ * CKV_;
  float* qcb = p; p += S_ * CQ_;
  float* qb = p;  p += S_ * NHHD_;
  float* kb = p;  p += S_ * NHHD_;
  float* vb = p;  p += S_ * NHHD_;
  float* kT = p;  p += (size_t)NH_ * HD_ * S_;
  float* sc = p;  p += (size_t)NH_ * S_ * S_;
  float* ao = p;  p += S_ * NHHD_;
  float* f1 = p;  p += (size_t)S_ * F_;
  float* wtok = p; p += S_ * E_;

  k_embed<<<S_, 256, 0, stream>>>(ids, emb, x0);

  float* xa = x0; float* xb = x1; float* xc = x2;
  for (int l = 0; l < L_; ++l) {
    k_ln<<<S_, 256, 0, stream>>>(xa, ln1g + l * H_, ln1b + l * H_, hb, H_);
    // kv = h @ Wdkv + bdkv
    GEMM(hb, H_, 0, Wdkv + (size_t)l * H_ * CKV_, CKV_, 0, bdkv + l * CKV_,
         nullptr, nullptr, 0, kvb, CKV_, 0, S_, CKV_, H_, 1, 1.f, 0, stream);
    // k = kv @ Wuk + buk ; v = kv @ Wuv + buv
    GEMM(kvb, CKV_, 0, Wuk + (size_t)l * CKV_ * NHHD_, NHHD_, 0, buk + l * NHHD_,
         nullptr, nullptr, 0, kb, NHHD_, 0, S_, NHHD_, CKV_, 1, 1.f, 0, stream);
    GEMM(kvb, CKV_, 0, Wuv + (size_t)l * CKV_ * NHHD_, NHHD_, 0, buv + l * NHHD_,
         nullptr, nullptr, 0, vb, NHHD_, 0, S_, NHHD_, CKV_, 1, 1.f, 0, stream);
    // q = (h @ Wdq + bdq) @ Wuq + buq
    GEMM(hb, H_, 0, Wdq + (size_t)l * H_ * CQ_, CQ_, 0, bdq + l * CQ_,
         nullptr, nullptr, 0, qcb, CQ_, 0, S_, CQ_, H_, 1, 1.f, 0, stream);
    GEMM(qcb, CQ_, 0, Wuq + (size_t)l * CQ_ * NHHD_, NHHD_, 0, buq + l * NHHD_,
         nullptr, nullptr, 0, qb, NHHD_, 0, S_, NHHD_, CQ_, 1, 1.f, 0, stream);
    // RoPE(q) in place; RoPE(k) -> kT [NH][HD][S]
    k_rope<<<(S_ * NH_ * HD2_) / 256, 256, 0, stream>>>(qb, kb, kT);
    // scores[h] = (q_h @ kT_h) / sqrt(HD), batched over heads
    GEMM(qb, NHHD_, HD_, kT, S_, (long long)HD_ * S_, nullptr, nullptr, nullptr, 0,
         sc, S_, (long long)S_ * S_, S_, S_, HD_, NH_, 0.125f, 0, stream);
    k_softmax<<<dim3(S_, NH_), 256, 0, stream>>>(sc);
    // ao[h] = P_h @ V_h, batched over heads
    GEMM(sc, S_, (long long)S_ * S_, vb, NHHD_, HD_, nullptr, nullptr, nullptr, 0,
         ao, NHHD_, HD_, S_, HD_, S_, NH_, 1.f, 0, stream);
    // xb = xa + ao @ Wo + bo
    GEMM(ao, NHHD_, 0, Wo + (size_t)l * NHHD_ * H_, H_, 0, bo + l * H_,
         xa, nullptr, 0, xb, H_, 0, S_, H_, NHHD_, 1, 1.f, 0, stream);
    // hm = ln2(xb); gate weights
    k_ln<<<S_, 256, 0, stream>>>(xb, ln2g + l * H_, ln2b + l * H_, hb, H_);
    k_gate<<<S_, 256, 0, stream>>>(hb, gateW + (size_t)l * H_ * E_,
                                   gateb + l * E_, gatebias + l * E_, wtok);
    // shared FFN: xc = xb + gelu(hm @ sW1 + sb1) @ sW2 + sb2
    GEMM(hb, H_, 0, sW1 + (size_t)l * H_ * F_, F_, 0, sb1 + l * F_,
         nullptr, nullptr, 0, f1, F_, 0, S_, F_, H_, 1, 1.f, GF_GELU, stream);
    GEMM(f1, F_, 0, sW2 + (size_t)l * F_ * H_, H_, 0, sb2 + l * H_,
         xb, nullptr, 0, xc, H_, 0, S_, H_, F_, 1, 1.f, 0, stream);
    // experts (dense, top-2 weighted accumulate into xc)
    for (int e = 0; e < E_; ++e) {
      GEMM(hb, H_, 0, eW1 + ((size_t)(l * E_ + e)) * H_ * F_, F_, 0,
           eb1 + (size_t)(l * E_ + e) * F_, nullptr, nullptr, 0,
           f1, F_, 0, S_, F_, H_, 1, 1.f, GF_GELU, stream);
      GEMM(f1, F_, 0, eW2 + ((size_t)(l * E_ + e)) * F_ * H_, H_, 0,
           eb2 + (size_t)(l * E_ + e) * H_, nullptr, wtok + e, E_,
           xc, H_, 0, S_, H_, F_, 1, 1.f, GF_ACC, stream);
    }
    float* t = xa; xa = xc; xc = t;
  }

  // final LN + LM head
  k_ln<<<S_, 256, 0, stream>>>(xa, flng, flnb, hb, H_);
  GEMM(hb, H_, 0, outW, V_, 0, outb, nullptr, nullptr, 0,
       out, V_, 0, S_, V_, H_, 1, 1.f, 0, stream);
}
